// DecoderBlock_82660940579418
// MI455X (gfx1250) — matrix operand order repaired and checked by reading, not compiled, not run on hardware
//
#include <hip/hip_runtime.h>
#include <hip/hip_bf16.h>
#include <cstdint>
#include <cstddef>

// ---------------------------------------------------------------------------
// DecoderBlock for MI455X (gfx1250): bf16 WMMA GEMMs fed by the Tensor Data
// Mover (A tiles) + async global->LDS copies (B tiles), double-buffered.
// Column-softmax + fused add+LN for the pointwise stages.
// ---------------------------------------------------------------------------

typedef __bf16 bf16_t;
typedef __attribute__((ext_vector_type(16))) __bf16 bf16x16;
typedef __attribute__((ext_vector_type(8)))  float  f32x8;
typedef unsigned int u32x4 __attribute__((ext_vector_type(4)));
typedef int          i32x8 __attribute__((ext_vector_type(8)));
typedef int          i32x4 __attribute__((ext_vector_type(4)));

#if __has_builtin(__builtin_amdgcn_tensor_load_to_lds)
#define ATHENA_HAVE_TDM 1
#else
#define ATHENA_HAVE_TDM 0
#endif

// Problem dims (fixed by the reference)
constexpr int Bc = 2, Sc = 1024, Dc = 2048, Hc = 16, DHc = 128, DFFc = 8192;
constexpr size_t BSD  = (size_t)Bc * Sc * Dc;       // 4M elems
constexpr size_t BHSE = (size_t)Bc * Hc * Sc * DHc; // 4M elems
constexpr size_t BHSS = (size_t)Bc * Hc * Sc * Sc;  // 32M elems

// GEMM tiling: block = 256 threads = 8 wave32; block tile 128x128, K-step 32.
// Each wave owns a 32x64 sub-tile = 2x4 WMMA 16x16 accumulators.
constexpr int BM = 128, BN = 128, BK = 32;
constexpr int LDK = BK + 8; // LDS k-stride: 64B data + 16B pad = 80B rows

// ---- gfx1250 async global->LDS copy (ASYNCcnt-tracked), inline asm --------
__device__ __forceinline__ void async_copy_b128(const void* gsrc, void* ldst)
{
    uint64_t ga = (uint64_t)(uintptr_t)gsrc;
    uint32_t la = (uint32_t)(uintptr_t)ldst; // generic LDS addr low 32 = LDS offset
    asm volatile("global_load_async_to_lds_b128 %0, %1, off"
                 :: "v"(la), "v"(ga) : "memory");
}
__device__ __forceinline__ void wait_asynccnt_0() { asm volatile("s_wait_asynccnt 0x0" ::: "memory"); }
__device__ __forceinline__ void wait_asynccnt_2() { asm volatile("s_wait_asynccnt 0x2" ::: "memory"); }
__device__ __forceinline__ void wait_asynccnt_4() { asm volatile("s_wait_asynccnt 0x4" ::: "memory"); }
__device__ __forceinline__ void wait_tensorcnt_0() { asm volatile("s_wait_tensorcnt 0x0" ::: "memory"); }
__device__ __forceinline__ void wait_tensorcnt_1() { asm volatile("s_wait_tensorcnt 0x1" ::: "memory"); }

#if ATHENA_HAVE_TDM
// ---- Tensor Data Mover: load a [nrows x BK] bf16 tile (row stride = lda
// elems) into LDS with 64B rows padded to 80B (matches LDK=40). One wave
// issues one tensor_load_to_lds; completion tracked with TENSORcnt.
// D# layout per CDNA5 ISA ch.8 (group0 128b, group1 256b; groups 2/3 unused).
// This toolchain exposes the 6-arg builtin form (extra int32x8 group + cpol).
__device__ __forceinline__ void tdm_load_tile(const void* gsrc, void* ldst,
                                              int lda, int nrows)
{
    uint64_t ga = (uint64_t)(uintptr_t)gsrc;
    uint32_t la = (uint32_t)(uintptr_t)ldst;
    u32x4 g0;
    g0[0] = 1u;                   // count=1 (valid user descriptor)
    g0[1] = la;                   // lds_addr (bytes)
    g0[2] = (uint32_t)ga;         // global_addr[31:0]
    g0[3] = (uint32_t)((ga >> 32) & 0x1FFFFFFu) | 0x80000000u; // [56:32] | type=2

    const unsigned td0 = (unsigned)lda;   // tensor dim0 (elems)
    const unsigned td1 = 1u << 20;        // tensor dim1 (rows), generous
    i32x8 g1;
    // data_size=1 (2B) | pad_enable | pad_interval=3 (16 DW = 64B)
    // | pad_amount=3 (4 DW = 16B); workgroup_mask=0 (not in a cluster)
    g1[0] = (1 << 16) | (1 << 20) | (3 << 22) | (3 << 25);
    g1[1] = (int)((td0 & 0xFFFFu) << 16);                 // atomic_bar=0 | td0.lo
    g1[2] = (int)((td0 >> 16) | ((td1 & 0xFFFFu) << 16)); // td0.hi | td1.lo
    g1[3] = (int)((td1 >> 16) | ((unsigned)BK << 16));    // td1.hi | tile_dim0
    g1[4] = (int)(nrows & 0xFFFF);                        // tile_dim1 | tile_dim2=0
    g1[5] = (int)lda;                                     // dim0_stride[31:0]
    g1[6] = 0;                                            // stride hi | dim1_stride.lo
    g1[7] = 0;
    const i32x4 gz4 = {0, 0, 0, 0};
    const i32x8 gz8 = {0, 0, 0, 0, 0, 0, 0, 0};
    __builtin_amdgcn_tensor_load_to_lds(g0, g1, gz4, gz4, gz8, 0);
}
#endif

// ---------------------------------------------------------------------------
// Generic batched GEMM:  C[z] = alpha * A[z] x B[z] + bias  (optional ReLU)
//   A: [M,K] bf16 row-major (TDM tile feed).
//   B: [N,K] bf16 (BT=1, async-to-LDS feed) or [K,N] bf16 (BT=0, scatter).
//   Batch z = bb*Hdim + hh with independent (bb,hh) strides per operand.
//   C written fp32 or bf16.  Double-buffered LDS pipeline.
// ---------------------------------------------------------------------------
template <bool BT, bool RELU, bool OUT_BF16>
__global__ __launch_bounds__(256) void gemm_wmma_kernel(
    const bf16_t* __restrict__ A, int lda, long long sAb, long long sAh,
    const bf16_t* __restrict__ Bm, int ldb, long long sBb, long long sBh,
    void* __restrict__ Cv, int ldc, long long sCb, long long sCh,
    const float* __restrict__ bias, int sBiasH,
    int M, int N, int K, int Hdim, float alpha)
{
    __shared__ bf16_t As[2][BM * LDK];
    __shared__ bf16_t Bs[2][BN * LDK];

    const int tid = threadIdx.x;
    const int z  = blockIdx.z;
    const int bb = z / Hdim, hh = z % Hdim;
    A  += (size_t)bb * sAb + (size_t)hh * sAh;
    Bm += (size_t)bb * sBb + (size_t)hh * sBh;
    const size_t cOff = (size_t)bb * sCb + (size_t)hh * sCh;
    float*  Cf = (float*)Cv  + cOff;
    bf16_t* Ch = (bf16_t*)Cv + cOff;
    const float* biasp = bias ? (bias + (size_t)hh * sBiasH) : nullptr;

    const int bm0 = blockIdx.y * BM;
    const int bn0 = blockIdx.x * BN;

    const int lane = tid & 31;
    const int wave = tid >> 5;   // 0..7
    const int wm   = wave >> 1;  // 0..3 -> row offset wm*32
    const int wn   = wave & 1;   // 0..1 -> col offset wn*64
    const int l15  = lane & 15;
    const int hi   = lane >> 4;  // half-wave select per ISA fragment layout

    // per-thread tile-feed coordinates
    const int ar = tid >> 2, acg = (tid & 3) << 3;           // 4 threads/row
    const int kk = tid >> 4, ng  = (tid & 15) << 3;          // B scatter coords

    auto loadA = [&](int kt, int buf) {
#if ATHENA_HAVE_TDM
        if (tid < 32) // wave 0 drives the TDM; EXEC-independent, 1 op per tile
            tdm_load_tile(A + (size_t)bm0 * lda + kt, &As[buf][0], lda, BM);
#else
#pragma unroll
        for (int c = 0; c < 2; c++) {
            int r = ar + c * 64;
            async_copy_b128(A + (size_t)(bm0 + r) * lda + kt + acg,
                            &As[buf][r * LDK + acg]);
        }
#endif
    };
    auto loadB = [&](int kt, int buf) {
        if (BT) { // B as [N,K]: contiguous along k -> pure async 16B copies
#pragma unroll
            for (int c = 0; c < 2; c++) {
                int n = ar + c * 64;
                async_copy_b128(Bm + (size_t)(bn0 + n) * ldb + kt + acg,
                                &Bs[buf][n * LDK + acg]);
            }
        } else { // B as [K,N]: coalesced read, 16-bit scatter-transpose to LDS
#pragma unroll
            for (int c = 0; c < 2; c++) {
                int k2 = kk + c * 16;
                union { uint4 q; bf16_t h[8]; } u;
                u.q = *(const uint4*)(Bm + (size_t)(kt + k2) * ldb + bn0 + ng);
#pragma unroll
                for (int i = 0; i < 8; i++) Bs[buf][(ng + i) * LDK + k2] = u.h[i];
            }
        }
    };

    f32x8 acc[2][4];
    const f32x8 zf = {0.f, 0.f, 0.f, 0.f, 0.f, 0.f, 0.f, 0.f};
#pragma unroll
    for (int i = 0; i < 2; i++)
#pragma unroll
        for (int j = 0; j < 4; j++) acc[i][j] = zf;

    const int nt = K / BK;
    loadA(0, 0);
    loadB(0, 0);

    for (int t = 0; t < nt; ++t) {
        const int cur = t & 1;
        const bool more = (t + 1 < nt);
        if (more) { loadA((t + 1) * BK, cur ^ 1); loadB((t + 1) * BK, cur ^ 1); }
        // TDM ops / async ops each complete in order within a wave: waiting
        // down to the NEXT tile's in-flight count guarantees THIS tile landed.
#if ATHENA_HAVE_TDM
        if (tid < 32) { if (more) wait_tensorcnt_1(); else wait_tensorcnt_0(); }
        if (BT) { if (more) wait_asynccnt_2(); else wait_asynccnt_0(); }
#else
        if (more) { if (BT) wait_asynccnt_4(); else wait_asynccnt_2(); }
        else      { wait_asynccnt_0(); }
#endif
        __syncthreads();

        // ---- fragments in contiguous k order (k = 16*hi + s), same for A and B -----------------------------
        // A 16x32: lane m=l15; lanes<16 K 0..15; lanes>=16 K 16..31
        // B 32x16: lane n=l15; lanes<16 K pairs 0..15; lanes>=16 K 16..31
        union { bf16x16 v; uint32_t u[8]; } af[2], bfm[4];
#pragma unroll
        for (int i = 0; i < 2; i++) {
            const bf16_t* ap = &As[cur][(wm * 32 + i * 16 + l15) * LDK];
            const int kb = hi * 16;
#pragma unroll
            for (int j = 0; j < 4; j++) af[i].u[j]     = *(const uint32_t*)(ap + kb + 2 * j);
#pragma unroll
            for (int j = 0; j < 4; j++) af[i].u[4 + j] = *(const uint32_t*)(ap + 8 + kb + 2 * j);
        }
#pragma unroll
        for (int jn = 0; jn < 4; jn++) {
            const bf16_t* bp = &Bs[cur][(wn * 64 + jn * 16 + l15) * LDK + hi * 16];
#pragma unroll
            for (int j = 0; j < 8; j++) bfm[jn].u[j] = *(const uint32_t*)(bp + 2 * j);
        }
#pragma unroll
        for (int i = 0; i < 2; i++)
#pragma unroll
            for (int jn = 0; jn < 4; jn++)
                acc[i][jn] = __builtin_amdgcn_wmma_f32_16x16x32_bf16(
                    false, af[i].v, false, bfm[jn].v, (short)0, acc[i][jn],
                    false, false);
        __syncthreads();
    }

    // ---- epilogue: C/D layout -> VGPR r holds M = r + 8*hi ----------------
#pragma unroll
    for (int i = 0; i < 2; i++) {
        int rowb = bm0 + wm * 32 + i * 16 + hi * 8;
#pragma unroll
        for (int jn = 0; jn < 4; jn++) {
            int col = bn0 + wn * 64 + jn * 16 + l15;
            float bv = biasp ? biasp[col] : 0.f;
#pragma unroll
            for (int r = 0; r < 8; r++) {
                float v = alpha * acc[i][jn][r] + bv;
                if (RELU) v = v > 0.f ? v : 0.f;
                size_t idx = (size_t)(rowb + r) * ldc + col;
                if (OUT_BF16) Ch[idx] = (bf16_t)v;
                else          Cf[idx] = v;
            }
        }
    }
}

// ---------------------------------------------------------------------------
// fp32 -> bf16 cast (grid-stride)
// ---------------------------------------------------------------------------
__global__ __launch_bounds__(256) void cast_bf16_kernel(
    const float* __restrict__ in, bf16_t* __restrict__ out, size_t n)
{
    size_t i = (size_t)blockIdx.x * blockDim.x + threadIdx.x;
    size_t stride = (size_t)gridDim.x * blockDim.x;
    for (; i < n; i += stride) out[i] = (bf16_t)in[i];
}

// fp32 [batch,K,N] -> bf16 transposed [batch,N,K]  (weights for BT GEMM path)
__global__ __launch_bounds__(256) void castT_bf16_kernel(
    const float* __restrict__ in, bf16_t* __restrict__ out,
    int Kr, int Ncols, int batch)
{
    size_t total = (size_t)batch * Kr * Ncols;
    size_t i = (size_t)blockIdx.x * blockDim.x + threadIdx.x;
    size_t stride = (size_t)gridDim.x * blockDim.x;
    for (; i < total; i += stride) {
        size_t h = i / ((size_t)Kr * Ncols);
        size_t r = i % ((size_t)Kr * Ncols);
        size_t k = r / Ncols, n = r % Ncols;
        out[h * (size_t)Kr * Ncols + n * Kr + k] = (bf16_t)in[i];
    }
}

// ---------------------------------------------------------------------------
// Softmax over the QUERY axis (reference: softmax(dim=1) -> axis=2 here):
// one thread per k column; row-q reads are contiguous across threads.
// ---------------------------------------------------------------------------
__global__ __launch_bounds__(256) void softmax_q_kernel(
    const float* __restrict__ scores, bf16_t* __restrict__ attn)
{
    const int S = Sc;
    const size_t bh = blockIdx.y;
    const int k = blockIdx.x * 256 + threadIdx.x;
    const float* base = scores + bh * (size_t)S * S + k;
    bf16_t* ob = attn + bh * (size_t)S * S + k;

    float mx = -3.4e38f;
    for (int q = 0; q < S; q++) mx = fmaxf(mx, base[(size_t)q * S]);
    float sum = 0.f;
    for (int q = 0; q < S; q++) sum += __expf(base[(size_t)q * S] - mx);
    float inv = 1.f / sum;
    for (int q = 0; q < S; q++)
        ob[(size_t)q * S] = (bf16_t)(__expf(base[(size_t)q * S] - mx) * inv);
}

// ---------------------------------------------------------------------------
// y = LayerNorm(x + m) * g + b ; writes fp32 (residual stream) and optional
// bf16 copy (next GEMM's A operand). One block per row of D=2048.
// ---------------------------------------------------------------------------
__global__ __launch_bounds__(256) void add_ln_kernel(
    const float* __restrict__ x, const float* __restrict__ m,
    const float* __restrict__ g, const float* __restrict__ b,
    float* __restrict__ outf, bf16_t* __restrict__ outb)
{
    const int D = Dc;
    const size_t row = blockIdx.x;
    const float* xr = x + row * D;
    const float* mr = m + row * D;
    __shared__ float red[256];

    float v[8];
    float s = 0.f;
#pragma unroll
    for (int i = 0; i < 8; i++) {
        int idx = threadIdx.x + i * 256;
        v[i] = xr[idx] + mr[idx];
        s += v[i];
    }
    red[threadIdx.x] = s;
    __syncthreads();
    for (int off = 128; off > 0; off >>= 1) {
        if (threadIdx.x < off) red[threadIdx.x] += red[threadIdx.x + off];
        __syncthreads();
    }
    float mean = red[0] / D;
    __syncthreads();

    float vs = 0.f;
#pragma unroll
    for (int i = 0; i < 8; i++) {
        float d = v[i] - mean;
        vs += d * d;
    }
    red[threadIdx.x] = vs;
    __syncthreads();
    for (int off = 128; off > 0; off >>= 1) {
        if (threadIdx.x < off) red[threadIdx.x] += red[threadIdx.x + off];
        __syncthreads();
    }
    float inv = rsqrtf(red[0] / D + 1e-5f);

#pragma unroll
    for (int i = 0; i < 8; i++) {
        int idx = threadIdx.x + i * 256;
        float y = (v[i] - mean) * inv * g[idx] + b[idx];
        outf[row * D + idx] = y;
        if (outb) outb[row * D + idx] = (bf16_t)y;
    }
}

// ---------------------------------------------------------------------------
// Host orchestration
// ---------------------------------------------------------------------------
static inline void cast_launch(const float* src, bf16_t* dst, size_t n, hipStream_t s)
{
    cast_bf16_kernel<<<2048, 256, 0, s>>>(src, dst, n);
}
static inline void castT_launch(const float* src, bf16_t* dst, int Kr, int Ncols,
                                int batch, hipStream_t s)
{
    castT_bf16_kernel<<<2048, 256, 0, s>>>(src, dst, Kr, Ncols, batch);
}

extern "C" void kernel_launch(void* const* d_in, const int* in_sizes, int n_in,
                              void* d_out, int out_size, void* d_ws, size_t ws_size,
                              hipStream_t stream)
{
    (void)in_sizes; (void)n_in; (void)out_size; (void)ws_size;
    typedef long long ll;

    const float* x    = (const float*)d_in[0];
    const float* enc  = (const float*)d_in[1];
    const float* wq1  = (const float*)d_in[2];
    const float* wk1  = (const float*)d_in[3];
    const float* wv1  = (const float*)d_in[4];
    const float* bq1  = (const float*)d_in[5];
    const float* bk1  = (const float*)d_in[6];
    const float* bv1  = (const float*)d_in[7];
    const float* wp1  = (const float*)d_in[8];
    const float* bp1  = (const float*)d_in[9];
    const float* wq2  = (const float*)d_in[10];
    const float* wk2  = (const float*)d_in[11];
    const float* wv2  = (const float*)d_in[12];
    const float* bq2  = (const float*)d_in[13];
    const float* bk2  = (const float*)d_in[14];
    const float* bv2  = (const float*)d_in[15];
    const float* wp2  = (const float*)d_in[16];
    const float* bp2  = (const float*)d_in[17];
    const float* ln1g = (const float*)d_in[18];
    const float* ln1b = (const float*)d_in[19];
    const float* ln2g = (const float*)d_in[20];
    const float* ln2b = (const float*)d_in[21];
    const float* ln3g = (const float*)d_in[22];
    const float* ln3b = (const float*)d_in[23];
    const float* wff1 = (const float*)d_in[24];
    const float* bff1 = (const float*)d_in[25];
    const float* wff2 = (const float*)d_in[26];
    const float* bff2 = (const float*)d_in[27];

    // ---- workspace carve-up ----
    char* w = (char*)d_ws;
    size_t off = 0;
    auto carve = [&](size_t bytes) -> char* {
        char* p = w + off;
        off += (bytes + 255) & ~(size_t)255;
        return p;
    };
    bf16_t* XB      = (bf16_t*)carve(BSD * 2);                 // x in bf16
    bf16_t* EB      = (bf16_t*)carve(BSD * 2);                 // encoder_output bf16
    bf16_t* QB      = (bf16_t*)carve(BHSE * 2);
    bf16_t* KB      = (bf16_t*)carve(BHSE * 2);
    bf16_t* VB      = (bf16_t*)carve(BHSE * 2);
    bf16_t* CONCATB = (bf16_t*)carve(BSD * 2);                 // head-concat, bf16
    float*  LN1F    = (float*) carve(BSD * 4);
    bf16_t* LN1B    = (bf16_t*)carve(BSD * 2);
    float*  LN2F    = (float*) carve(BSD * 4);
    bf16_t* LN2B    = (bf16_t*)carve(BSD * 2);
    float*  MF      = (float*) carve(BSD * 4);                 // proj / ffn output
    bf16_t* HIDB    = (bf16_t*)carve((size_t)Bc * Sc * DFFc * 2);
    bf16_t* WSCR    = (bf16_t*)carve((size_t)Dc * DFFc * 2);   // W^T bf16 scratch
    float*  SCORES  = (float*) carve(BHSS * 4);
    bf16_t* ATTNB   = (bf16_t*)carve(BHSS * 2);

    const float invSqrtDH = 0.088388347648318447f; // 1/sqrt(128)

    cast_launch(x, XB, BSD, stream);
    cast_launch(enc, EB, BSD, stream);

    // ===== one attention block ============================================
    auto attention = [&](const bf16_t* qin, const bf16_t* kvin,
                         const float* wq, const float* wk, const float* wv,
                         const float* bq, const float* bk, const float* bv,
                         const float* wp, const float* bp) {
        dim3 gQKV(DHc / BN, Sc / BM, Bc * Hc);
        // Q = qin x wq[h] + bq[h]  -> [B,H,S,DH] bf16. Weights pre-transposed
        // per head to [DH, D] so the GEMM's B feed is pure async b128 copies.
        castT_launch(wq, WSCR, Dc, DHc, Hc, stream);
        gemm_wmma_kernel<true, false, true><<<gQKV, 256, 0, stream>>>(
            qin, Dc, (ll)Sc * Dc, 0, WSCR, Dc, 0, (ll)Dc * DHc,
            QB, DHc, (ll)Hc * Sc * DHc, (ll)Sc * DHc,
            bq, DHc, Sc, DHc, Dc, Hc, 1.0f);
        castT_launch(wk, WSCR, Dc, DHc, Hc, stream);
        gemm_wmma_kernel<true, false, true><<<gQKV, 256, 0, stream>>>(
            kvin, Dc, (ll)Sc * Dc, 0, WSCR, Dc, 0, (ll)Dc * DHc,
            KB, DHc, (ll)Hc * Sc * DHc, (ll)Sc * DHc,
            bk, DHc, Sc, DHc, Dc, Hc, 1.0f);
        castT_launch(wv, WSCR, Dc, DHc, Hc, stream);
        gemm_wmma_kernel<true, false, true><<<gQKV, 256, 0, stream>>>(
            kvin, Dc, (ll)Sc * Dc, 0, WSCR, Dc, 0, (ll)Dc * DHc,
            VB, DHc, (ll)Hc * Sc * DHc, (ll)Sc * DHc,
            bv, DHc, Sc, DHc, Dc, Hc, 1.0f);

        // scores = Q x K^T / sqrt(DH): K is [S,DH] == [N,K] -> BT path, fp32 out
        dim3 gS(Sc / BN, Sc / BM, Bc * Hc);
        gemm_wmma_kernel<true, false, false><<<gS, 256, 0, stream>>>(
            QB, DHc, (ll)Hc * Sc * DHc, (ll)Sc * DHc,
            KB, DHc, (ll)Hc * Sc * DHc, (ll)Sc * DHc,
            SCORES, Sc, (ll)Hc * Sc * Sc, (ll)Sc * Sc,
            nullptr, 0, Sc, Sc, DHc, Hc, invSqrtDH);

        // softmax over q (axis=2 in reference) -> bf16 attn
        softmax_q_kernel<<<dim3(Sc / 256, Bc * Hc), 256, 0, stream>>>(SCORES, ATTNB);

        // out = attn x V, written straight into [B,S,H*DH] concat layout.
        // V is [S,DH] == [K,N] -> scatter-transpose B path.
        dim3 gAV(DHc / BN, Sc / BM, Bc * Hc);
        gemm_wmma_kernel<false, false, true><<<gAV, 256, 0, stream>>>(
            ATTNB, Sc, (ll)Hc * Sc * Sc, (ll)Sc * Sc,
            VB, DHc, (ll)Hc * Sc * DHc, (ll)Sc * DHc,
            CONCATB, Dc, (ll)Sc * Dc, (ll)DHc,
            nullptr, 0, Sc, DHc, Sc, Hc, 1.0f);

        // projection: concat x wp + bp -> MF fp32  (wp pre-transposed)
        castT_launch(wp, WSCR, Dc, Dc, 1, stream);
        dim3 gP(Dc / BN, (Bc * Sc) / BM, 1);
        gemm_wmma_kernel<true, false, false><<<gP, 256, 0, stream>>>(
            CONCATB, Dc, 0, 0, WSCR, Dc, 0, 0, MF, Dc, 0, 0,
            bp, 0, Bc * Sc, Dc, Dc, 1, 1.0f);
    };

    // ===== self-attention + LN1 ===========================================
    attention(XB, XB, wq1, wk1, wv1, bq1, bk1, bv1, wp1, bp1);
    add_ln_kernel<<<Bc * Sc, 256, 0, stream>>>(x, MF, ln1g, ln1b, LN1F, LN1B);

    // ===== cross-attention (query = encoder_output, kv = x) + LN2 =========
    attention(EB, LN1B, wq2, wk2, wv2, bq2, bk2, bv2, wp2, bp2);
    add_ln_kernel<<<Bc * Sc, 256, 0, stream>>>(LN1F, MF, ln2g, ln2b, LN2F, LN2B);

    // ===== FFN + LN3 ======================================================
    castT_launch(wff1, WSCR, Dc, DFFc, 1, stream);   // [D,DFF] -> [DFF,D]
    dim3 gF1(DFFc / BN, (Bc * Sc) / BM, 1);
    gemm_wmma_kernel<true, true, true><<<gF1, 256, 0, stream>>>(
        LN2B, Dc, 0, 0, WSCR, Dc, 0, 0, HIDB, DFFc, 0, 0,
        bff1, 0, Bc * Sc, DFFc, Dc, 1, 1.0f);

    castT_launch(wff2, WSCR, DFFc, Dc, 1, stream);   // [DFF,D] -> [D,DFF]
    dim3 gF2(Dc / BN, (Bc * Sc) / BM, 1);
    gemm_wmma_kernel<true, false, false><<<gF2, 256, 0, stream>>>(
        HIDB, DFFc, 0, 0, WSCR, DFFc, 0, 0, MF, Dc, 0, 0,
        bff2, 0, Bc * Sc, Dc, DFFc, 1, 1.0f);

    add_ln_kernel<<<Bc * Sc, 256, 0, stream>>>(LN2F, MF, ln3g, ln3b,
                                               (float*)d_out, nullptr);
}
